// MPMCRN_63127429316725
// MI455X (gfx1250) — matrix operand order repaired and checked by reading, not compiled, not run on hardware
//
#include <hip/hip_runtime.h>
#include <hip/hip_bf16.h>

// MI455X / gfx1250, wave32. bf16 WMMA (v_wmma_f32_16x16x32_bf16) with f32 accum.
// Scan stages Wh in LDS via the Tensor Data Mover (tensor_load_to_lds + s_wait_tensorcnt),
// double-buffers the h tile in LDS, and prefetches next-step GX (global_prefetch_b8).

typedef __attribute__((ext_vector_type(16))) __bf16 v16bf;
typedef __attribute__((ext_vector_type(8)))  float  v8f;
typedef __attribute__((ext_vector_type(4)))  unsigned int u32x4;
typedef __attribute__((ext_vector_type(8)))  int i32x8;
typedef __attribute__((ext_vector_type(4)))  int i32x4;

#define B_SZ 256
#define T_SZ 50
#define V_SZ 100000
#define D_SZ 128
#define H_SZ 128
#define GXW  1152     // 3 cells * 3H
#define REC  10
#define TEMP_INV 2.0f
#define THRESH 0.01f

#define WH_BYTES (GXW * 128 * 2)                 // 294912 B bf16 N-major weights
#define HB_BYTES (16 * H_SZ * 2)                 // 4 KB per h buffer
#define SCAN_LDS (WH_BYTES + 2 * HB_BYTES)       // 303104 B < 320 KB/WGP

union V16U { v16bf v; uint4 u[2]; };

__device__ __forceinline__ v8f wmma_bf16(v16bf a, v16bf b, v8f c) {
  return __builtin_amdgcn_wmma_f32_16x16x32_bf16(false, a, false, b, (short)0, c,
                                                 false, false);
}

// ---- Tensor Data Mover: 2D tile (rows x row_qwords*8B), contiguous, global -> LDS.
// D# packing per CDNA5 ISA:
//  g0: [1:0]count=1 | lds_addr[63:32] | global_addr[120:64] | type=2 @ [127:126]
//  g1: data_size=3(8B)@[17:16], tensor_dim0@[79:48], tensor_dim1@[111:80],
//      tile_dim0@[127:112], tile_dim1@[143:128], tensor_dim0_stride@[207:160]
__device__ __forceinline__ void tdm_load_to_lds(const void* gsrc, unsigned ldsOff,
                                                unsigned rows, unsigned row_qwords) {
  unsigned long long ga = (unsigned long long)(size_t)gsrc;
  u32x4 g0 = {1u, ldsOff, (unsigned)ga,
              (unsigned)((ga >> 32) & 0x1FFFFFFu) | (2u << 30)};
  i32x8 g1;
  g1[0] = (int)(3u << 16);                                   // data_size = 8 bytes
  g1[1] = (int)((row_qwords & 0xFFFFu) << 16);               // tensor_dim0[15:0]
  g1[2] = (int)((row_qwords >> 16) | ((rows & 0xFFFFu) << 16));  // dim0 hi | dim1 lo
  g1[3] = (int)((rows >> 16) | ((row_qwords & 0xFFFFu) << 16));  // dim1 hi | tile_dim0
  g1[4] = (int)(rows & 0xFFFFu);                             // tile_dim1 (tile_dim2=0)
  g1[5] = (int)row_qwords;                                   // tensor_dim0_stride lo
  g1[6] = 0;
  g1[7] = 0;
  i32x4 z4 = {0, 0, 0, 0};
#if __has_include(<hip/amd_detail/amd_gfx1250_TDM.h>)
  i32x8 z8 = {0, 0, 0, 0, 0, 0, 0, 0};
  __builtin_amdgcn_tensor_load_to_lds(g0, g1, z4, z4, z8, 0);
#else
  __builtin_amdgcn_tensor_load_to_lds(g0, g1, z4, z4, 0);
#endif
}

// ---- WMMA operand loaders (layouts per CDNA5 ISA §7.12.2) ----
// A (16x32 bf16, M x K): lane<16: M=lane, elems0-7 = K k0..k0+7, elems8-15 = k0+16..k0+23
//                        lane>=16: M=lane-16, K offset +8 / +24.
__device__ __forceinline__ v16bf load_A_g(const __bf16* __restrict__ A, int ld,
                                          int m0, int k0) {
  int lane = threadIdx.x & 31;
  const __bf16* p = A + (size_t)(m0 + (lane & 15)) * ld + k0 + ((lane >> 4) << 3);
  V16U t;
  t.u[0] = *(const uint4*)(p);
  t.u[1] = *(const uint4*)(p + 16);
  return t.v;
}
__device__ __forceinline__ v16bf load_A_lds(const __bf16* hsh, int k0) {
  int lane = threadIdx.x & 31;
  const __bf16* p = hsh + (lane & 15) * H_SZ + k0 + ((lane >> 4) << 3);
  V16U t;
  t.u[0] = *(const uint4*)(p);
  t.u[1] = *(const uint4*)(p + 16);
  return t.v;
}
// B (32x16 bf16, K x N), weight stored N-major (N x 128 bf16):
// lane<16: N=lane, elems0-7 = K k0..k0+7, elems8-15 = k0+16..k0+23; lane>=16: N=lane-16, K offset +8 / +24.
__device__ __forceinline__ v16bf load_B_nt(const __bf16* WT, int n0, int k0) {
  int lane = threadIdx.x & 31;
  const __bf16* p = WT + (size_t)(n0 + (lane & 15)) * 128 + k0 + ((lane >> 4) << 3);
  V16U t;
  t.u[0] = *(const uint4*)(p);
  t.u[1] = *(const uint4*)(p + 16);
  return t.v;
}
// C layout: elem v -> row = v + (lane>=16 ? 8 : 0), col = lane&15.
__device__ __forceinline__ v8f load_gx_tile(const float* __restrict__ GX, int b0,
                                            int t, int col0) {
  int lane = threadIdx.x & 31;
  int col = col0 + (lane & 15);
  int rb = (lane >> 4) << 3;
  v8f r;
#pragma unroll
  for (int v = 0; v < 8; ++v)
    r[v] = GX[(size_t)((b0 + rb + v) * T_SZ + t) * GXW + col];
  return r;
}

// ---------------- prep kernels ----------------
__global__ void k_emb_bf16(const float* __restrict__ emb, __bf16* __restrict__ eb) {
  size_t i = (size_t)blockIdx.x * 256 + threadIdx.x;   // 100000*128 threads
  if (i >= (size_t)V_SZ * D_SZ) return;
  float v = (i < D_SZ) ? 0.0f : emb[i];                // emb.at[0].set(0)
  eb[i] = (__bf16)v;
}
__global__ void k_w_bf16(const float* __restrict__ wx, const float* __restrict__ wh,
                         __bf16* __restrict__ wxt, __bf16* __restrict__ wht) {
  int i = blockIdx.x * 256 + threadIdx.x;              // 1152*128 threads
  if (i >= GXW * 128) return;
  int n = i >> 7, k = i & 127;                         // n in [0,1152)
  int j = n / 384, nl = n % 384;
  size_t src = (size_t)j * 128 * 384 + (size_t)k * 384 + nl;  // (3,128,384)
  wxt[(size_t)n * 128 + k] = (__bf16)wx[src];
  wht[(size_t)n * 128 + k] = (__bf16)wh[src];
}
__global__ void k_gather(const int* __restrict__ ix, const float* __restrict__ emb,
                         __bf16* __restrict__ xe) {
  int pos = blockIdx.x;                                // B*T blocks, 128 threads
  int d = threadIdx.x;
  int idx = ix[pos];
  float v = (idx == 0) ? 0.0f : emb[(size_t)idx * D_SZ + d];
  xe[(size_t)pos * D_SZ + d] = (__bf16)v;
}
__global__ void k_router(const __bf16* __restrict__ xe, const float* __restrict__ rw,
                         const float* __restrict__ rb, float* __restrict__ cout) {
  int pos = blockIdx.x * 256 + threadIdx.x;            // B*T threads
  if (pos >= B_SZ * T_SZ) return;
  float acc[3] = {rb[0], rb[1], rb[2]};
  const __bf16* x = xe + (size_t)pos * D_SZ;
  for (int k = 0; k < D_SZ; ++k) {
    float xv = (float)x[k];
    acc[0] += xv * rw[k * 3 + 0];
    acc[1] += xv * rw[k * 3 + 1];
    acc[2] += xv * rw[k * 3 + 2];
  }
  float m = fmaxf(acc[0], fmaxf(acc[1], acc[2]));
  float e0 = __expf((acc[0] - m) * TEMP_INV);
  float e1 = __expf((acc[1] - m) * TEMP_INV);
  float e2 = __expf((acc[2] - m) * TEMP_INV);
  float s = 1.0f / (e0 + e1 + e2);
  cout[pos * 3 + 0] = e0 * s;
  cout[pos * 3 + 1] = e1 * s;
  cout[pos * 3 + 2] = e2 * s;
}
__global__ void k_lengths(const int* __restrict__ ix, int* __restrict__ lens) {
  int b = threadIdx.x;                                 // 1 block x 256
  if (b >= B_SZ) return;
  int cnt = 0;
  for (int t = 0; t < T_SZ; ++t) cnt += (ix[b * T_SZ + t] != 0);
  int len = cnt - 1;
  if (len < 0) len = T_SZ - 1;                         // jnp index -1 wraps
  lens[b] = len;
}

// ---------------- GX = x_emb @ Wx_all + bias  (12800 x 128 x 1152) ----------------
__global__ void k_gx_gemm(const __bf16* __restrict__ xe, const __bf16* __restrict__ wxt,
                          const float* __restrict__ cb, float* __restrict__ GX) {
  int wid = blockIdx.x * 8 + (threadIdx.x >> 5);       // 57600 wave-tiles
  int mt = wid / 72, nt = wid % 72;
  int m0 = mt * 16, n0 = nt * 16;
  v8f acc = {};
#pragma unroll
  for (int k0 = 0; k0 < 128; k0 += 32) {
    v16bf a = load_A_g(xe, D_SZ, m0, k0);
    v16bf b = load_B_nt(wxt, n0, k0);
    acc = wmma_bf16(a, b, acc);
  }
  int lane = threadIdx.x & 31;
  int col = n0 + (lane & 15);
  float bias = cb[col];                                // cell_b flat (3,384) == col
  int rb = (lane >> 4) << 3;
#pragma unroll
  for (int v = 0; v < 8; ++v)
    GX[(size_t)(m0 + rb + v) * GXW + col] = acc[v] + bias;
}

// ---------------- recurrent scan: 16 batch rows per block, 8 waves ----------------
// Dynamic LDS: [0, 288KB) = Wh (all 3 cells, N-major bf16) staged once by the TDM;
//              then two 4 KB h buffers (double-buffered: read p, write p^1).
__global__ void k_scan(const float* __restrict__ GX, const __bf16* __restrict__ wht,
                       const float* __restrict__ call, const int* __restrict__ lens,
                       __bf16* __restrict__ ctxb) {
  extern __shared__ char dynsm[];
  __bf16* whLds = (__bf16*)dynsm;
  __bf16* hb0 = (__bf16*)(dynsm + WH_BYTES);
  __bf16* hb1 = (__bf16*)(dynsm + WH_BYTES + HB_BYTES);
  __shared__ float csh[3][16];
  __shared__ int lensh[16];
  int b0 = blockIdx.x * 16;
  int tid = threadIdx.x, lane = tid & 31, w = tid >> 5;
  if (tid < 32) {                                      // wave 0 drives the TDM
    unsigned ldsOff = (unsigned)(size_t)(void*)whLds;  // LDS byte offset (addr[31:0])
    tdm_load_to_lds(wht, ldsOff, GXW, (128 * 2) / 8);  // 1152 rows x 32 qwords
    __builtin_amdgcn_s_wait_tensorcnt(0);
  }
  for (int i = tid; i < 16 * H_SZ; i += 256) hb0[i] = (__bf16)0.0f;
  if (tid < 16) lensh[tid] = lens[b0 + tid];
  v8f hreg = {};
  v8f ctx = {};
  __syncthreads();
  int colbase = w * 16;                                // this wave's H-column slice
  int p = 0;
  for (int t = 0; t < T_SZ; ++t) {
    if (tid < 48) {
      int i = tid & 15, j = tid >> 4;
      csh[j][i] = call[((size_t)(b0 + i) * T_SZ + t) * 3 + j];
    }
    if (t + 1 < T_SZ) {                                // prefetch next step's GX rows
      int ri = tid >> 4, seg = tid & 15;               // 16 rows x 16 x 288B segments
      const float* pf = GX + (size_t)((b0 + ri) * T_SZ + t + 1) * GXW + seg * 72;
      __builtin_prefetch(pf, 0, 0);
      __builtin_prefetch(pf + 32, 0, 0);
      __builtin_prefetch(pf + 64, 0, 0);
    }
    __syncthreads();
#pragma unroll
    for (int j = 0; j < 3; ++j) {
      const __bf16* hrd = p ? hb1 : hb0;               // read h after cell j-1
      __bf16* hwr = p ? hb0 : hb1;                     // write h after cell j
      int colR = j * 384 + colbase;
      int colZ = colR + 128, colN = colR + 256;
      v8f accR = load_gx_tile(GX, b0, t, colR);        // init with gx (r,z)
      v8f accZ = load_gx_tile(GX, b0, t, colZ);
      v8f gxn  = load_gx_tile(GX, b0, t, colN);        // gx_n added post-r*gh_n
      v8f accN = {};
#pragma unroll
      for (int k0 = 0; k0 < 128; k0 += 32) {
        v16bf a = load_A_lds(hrd, k0);                 // A = h (shared by 3 tiles)
        accR = wmma_bf16(a, load_B_nt(whLds, colR, k0), accR);
        accZ = wmma_bf16(a, load_B_nt(whLds, colZ, k0), accZ);
        accN = wmma_bf16(a, load_B_nt(whLds, colN, k0), accN);
      }
      int rb = (lane >> 4) << 3;
#pragma unroll
      for (int v = 0; v < 8; ++v) {
        int row = rb + v;
        float r = 1.0f / (1.0f + __expf(-accR[v]));
        float z = 1.0f / (1.0f + __expf(-accZ[v]));
        float n = tanhf(gxn[v] + r * accN[v]);
        float cj = csh[j][row];
        float g = (cj > THRESH) ? cj * z : 0.0f;
        float hn = (1.0f - g) * hreg[v] + g * n;
        hreg[v] = hn;
        hwr[row * H_SZ + colbase + (lane & 15)] = (__bf16)hn;
        if (t == lensh[row]) ctx[v] += cj * hn;        // context accumulation
      }
      p ^= 1;
      __syncthreads();                                 // h(j) visible for cell j+1
    }
  }
  int rb = (lane >> 4) << 3, col = colbase + (lane & 15);
#pragma unroll
  for (int v = 0; v < 8; ++v)
    ctxb[(size_t)(b0 + rb + v) * H_SZ + col] = (__bf16)ctx[v];
}

// ---------------- logits = context @ emb.T  (256 x 128 x 100000) ----------------
__global__ void k_logits(const __bf16* __restrict__ ctxb, const __bf16* __restrict__ eb,
                         float* __restrict__ lg) {
  int wid = blockIdx.x * 8 + (threadIdx.x >> 5);       // 100000 wave-tiles
  int mt = wid / 6250, nt = wid % 6250;
  int m0 = mt * 16, n0 = nt * 16;
  v8f acc = {};
#pragma unroll
  for (int k0 = 0; k0 < 128; k0 += 32) {
    v16bf a = load_A_g(ctxb, H_SZ, m0, k0);
    v16bf b = load_B_nt(eb, n0, k0);                   // emb row-major == B N-major
    acc = wmma_bf16(a, b, acc);
  }
  int lane = threadIdx.x & 31;
  int col = n0 + (lane & 15);
  int rb = (lane >> 4) << 3;
#pragma unroll
  for (int v = 0; v < 8; ++v)
    lg[(size_t)(m0 + rb + v) * V_SZ + col] = acc[v];
}

// ---------------- top-10 per row ----------------
__global__ void k_topk(const float* __restrict__ lg, float* __restrict__ rec) {
  __shared__ float sv[256];
  __shared__ int si[256];
  __shared__ int wt;
  int b = blockIdx.x, tid = threadIdx.x;
  const float* row = lg + (size_t)b * V_SZ;
  float tv[REC];
  int ti[REC];
#pragma unroll
  for (int i = 0; i < REC; ++i) { tv[i] = -3.0e38f; ti[i] = 0; }
  for (int i = tid; i < V_SZ; i += 256) {
    float v = row[i];
    if (v > tv[REC - 1]) {
      int p = REC - 1;
      while (p > 0 && tv[p - 1] < v) { tv[p] = tv[p - 1]; ti[p] = ti[p - 1]; --p; }
      tv[p] = v; ti[p] = i;
    }
  }
  int ptr = 0;
  for (int k = 0; k < REC; ++k) {
    sv[tid] = (ptr < REC) ? tv[ptr] : -3.0e38f;
    si[tid] = tid;
    __syncthreads();
    for (int s = 128; s > 0; s >>= 1) {
      if (tid < s && sv[tid + s] > sv[tid]) { sv[tid] = sv[tid + s]; si[tid] = si[tid + s]; }
      __syncthreads();
    }
    if (tid == 0) wt = si[0];
    __syncthreads();
    if (tid == wt) { rec[b * REC + k] = (float)ti[ptr]; ++ptr; }
    __syncthreads();
  }
}

// ---------------- workspace layout (bytes, 256-aligned) ----------------
static constexpr size_t OFF_XE   = 0;                                  // bf16 12800x128
static constexpr size_t OFF_WXT  = OFF_XE  + (size_t)12800 * 128 * 2;
static constexpr size_t OFF_WHT  = OFF_WXT + (size_t)GXW * 128 * 2;
static constexpr size_t OFF_EMBB = OFF_WHT + (size_t)GXW * 128 * 2;
static constexpr size_t OFF_C    = OFF_EMBB + (size_t)V_SZ * D_SZ * 2;
static constexpr size_t OFF_LEN  = OFF_C + (size_t)12800 * 3 * 4;
static constexpr size_t OFF_GX   = OFF_LEN + 1024;
static constexpr size_t OFF_CTXB = OFF_GX + (size_t)12800 * GXW * 4;

extern "C" void kernel_launch(void* const* d_in, const int* in_sizes, int n_in,
                              void* d_out, int out_size, void* d_ws, size_t ws_size,
                              hipStream_t stream) {
  const int*   item_x   = (const int*)d_in[0];
  // d_in[1] = all_samples (unused by reference output)
  const float* emb      = (const float*)d_in[2];
  const float* router_w = (const float*)d_in[3];
  const float* router_b = (const float*)d_in[4];
  const float* cell_wx  = (const float*)d_in[5];
  const float* cell_wh  = (const float*)d_in[6];
  const float* cell_b   = (const float*)d_in[7];

  char* ws = (char*)d_ws;
  __bf16* xe   = (__bf16*)(ws + OFF_XE);
  __bf16* wxt  = (__bf16*)(ws + OFF_WXT);
  __bf16* wht  = (__bf16*)(ws + OFF_WHT);
  __bf16* eb   = (__bf16*)(ws + OFF_EMBB);
  float*  call = (float*)(ws + OFF_C);
  int*    lens = (int*)(ws + OFF_LEN);
  float*  GX   = (float*)(ws + OFF_GX);
  __bf16* ctxb = (__bf16*)(ws + OFF_CTXB);

  float* rec = (float*)d_out;                 // (256,10) recommended (as float)
  float* lg  = rec + (size_t)B_SZ * REC;      // (256,100000) item_logits

  k_emb_bf16<<<(V_SZ * D_SZ) / 256, 256, 0, stream>>>(emb, eb);
  k_w_bf16  <<<(GXW * 128) / 256, 256, 0, stream>>>(cell_wx, cell_wh, wxt, wht);
  k_gather  <<<B_SZ * T_SZ, 128, 0, stream>>>(item_x, emb, xe);
  k_router  <<<(B_SZ * T_SZ) / 256, 256, 0, stream>>>(xe, router_w, router_b, call);
  k_lengths <<<1, 256, 0, stream>>>(item_x, lens);
  k_gx_gemm <<<(800 * 72) / 8, 256, 0, stream>>>(xe, wxt, cell_b, GX);
  k_scan    <<<B_SZ / 16, 256, SCAN_LDS, stream>>>(GX, wht, call, lens, ctxb);
  k_logits  <<<(16 * 6250) / 8, 256, 0, stream>>>(ctxb, eb, lg);
  k_topk    <<<B_SZ, 256, 0, stream>>>(lg, rec);
}
